// AdapterLayer_88244398063757
// MI455X (gfx1250) — matrix operand order repaired and checked by reading, not compiled, not run on hardware
//
#include <hip/hip_runtime.h>
#include <hip/hip_bf16.h>
#include <math.h>

// ---------------------------------------------------------------------------
// Types for CDNA5 WMMA (wave32): v_wmma_f32_16x16x32_bf16
// ---------------------------------------------------------------------------
typedef __bf16 bh;
typedef __attribute__((ext_vector_type(16))) __bf16 v16bf;
typedef __attribute__((ext_vector_type(8)))  float  v8f;

union FragU { v16bf v; uint4 q[2]; };

// Load a 16-element bf16 fragment from two 16B-aligned 16-byte chunks.
__device__ __forceinline__ v16bf frag2(const bh* lo, const bh* hi) {
    FragU u;
    u.q[0] = *reinterpret_cast<const uint4*>(lo);
    u.q[1] = *reinterpret_cast<const uint4*>(hi);
    return u.v;
}

__device__ __forceinline__ v8f wmma_bf16(v16bf a, v16bf b, v8f c) {
    // (neg_a, A, neg_b, B, c_mod, C, reuse_a, reuse_b)
    return __builtin_amdgcn_wmma_f32_16x16x32_bf16(false, a, false, b,
                                                   (short)0, c, false, false);
}

__device__ __forceinline__ float gelu_exact(float v) {
    return 0.5f * v * (1.f + erff(v * 0.70710678118654752440f));
}

// ---------------------------------------------------------------------------
// Kernel 0: f32 -> bf16 weight conversion (weights stay resident in L2)
// ---------------------------------------------------------------------------
__global__ void f2bf_kernel(const float* __restrict__ s, bh* __restrict__ d, int n) {
    int i = blockIdx.x * blockDim.x + threadIdx.x;
    if (i < n) d[i] = (bh)s[i];
}

// ---------------------------------------------------------------------------
// Kernel 1: per-(b,c) stats — pooled = mean(x), h = mean(gelu(highpass(x)))
// One block per (b,c); 3x3 depthwise high-pass with zero padding.
// ---------------------------------------------------------------------------
__global__ __launch_bounds__(256) void stats_kernel(const float* __restrict__ x,
                                                    float* __restrict__ hbuf,
                                                    float* __restrict__ pbuf) {
    int b = blockIdx.x >> 8;
    int c = blockIdx.x & 255;
    const float* xc = x + ((size_t)(b * 256 + c)) * 16384;
    float sp = 0.f, shg = 0.f;
    for (int p = threadIdx.x; p < 16384; p += 256) {
        int i = p >> 7, j = p & 127;
        float ctr = xc[p];
        int im = i - 1, ip = i + 1, jm = j - 1, jp = j + 1;
        bool up = im >= 0, dn = ip < 128, lf = jm >= 0, rt = jp < 128;
        float ns = 0.f;
        if (up) {
            if (lf) ns += xc[im * 128 + jm];
            ns += xc[im * 128 + j];
            if (rt) ns += xc[im * 128 + jp];
        }
        if (lf) ns += xc[i * 128 + jm];
        if (rt) ns += xc[i * 128 + jp];
        if (dn) {
            if (lf) ns += xc[ip * 128 + jm];
            ns += xc[ip * 128 + j];
            if (rt) ns += xc[ip * 128 + jp];
        }
        float hp = 8.f * ctr - ns;
        shg += gelu_exact(hp);
        sp += ctr;
    }
    __shared__ float r1[256], r2[256];
    r1[threadIdx.x] = sp;
    r2[threadIdx.x] = shg;
    __syncthreads();
    for (int s2 = 128; s2 > 0; s2 >>= 1) {
        if (threadIdx.x < (unsigned)s2) {
            r1[threadIdx.x] += r1[threadIdx.x + s2];
            r2[threadIdx.x] += r2[threadIdx.x + s2];
        }
        __syncthreads();
    }
    if (threadIdx.x == 0) {
        pbuf[b * 256 + c] = r1[0] * (1.f / 16384.f);
        hbuf[b * 256 + c] = r2[0] * (1.f / 16384.f);
    }
}

// ---------------------------------------------------------------------------
// Kernel 2: routing — MLP(h), logits, softmax, top-2 -> (eidx, gval, gsum)
// Single block; tiny problem (8 rows).
// ---------------------------------------------------------------------------
__global__ __launch_bounds__(256) void route_kernel(
    const float* __restrict__ h, const float* __restrict__ pooled,
    const float* __restrict__ w1, const float* __restrict__ b1,
    const float* __restrict__ w2, const float* __restrict__ b2,
    const float* __restrict__ gw, const float* __restrict__ fgw,
    int* __restrict__ eidx, float* __restrict__ gval, float* __restrict__ gsum) {
    __shared__ float sh1[8 * 512];
    __shared__ float fe[8 * 256];
    __shared__ float lg[8 * 4];
    int tid = threadIdx.x;
    for (int idx = tid; idx < 8 * 512; idx += 256) {
        int bb = idx >> 9, o = idx & 511;
        const float* hv = h + bb * 256;
        const float* wr = w1 + o * 256;
        float acc = b1[o];
        for (int d = 0; d < 256; ++d) acc += hv[d] * wr[d];
        sh1[idx] = gelu_exact(acc);
    }
    __syncthreads();
    for (int idx = tid; idx < 8 * 256; idx += 256) {
        int bb = idx >> 8, o = idx & 255;
        const float* hv = sh1 + bb * 512;
        const float* wr = w2 + o * 512;
        float acc = b2[o];
        for (int d = 0; d < 512; ++d) acc += hv[d] * wr[d];
        fe[idx] = acc;
    }
    __syncthreads();
    if (tid < 32) {
        int bb = tid >> 2, e = tid & 3;
        const float* pv = pooled + bb * 256;
        const float* gr = gw + e * 256;
        const float* fv = fe + bb * 256;
        const float* fr = fgw + e * 256;
        float acc = 0.f;
        for (int d = 0; d < 256; ++d) acc += pv[d] * gr[d] + fv[d] * fr[d];
        lg[tid] = acc;
    }
    __syncthreads();
    if (tid < 8) {
        float g[4];
        float mx = -1e30f;
        for (int e = 0; e < 4; ++e) { g[e] = lg[tid * 4 + e]; mx = fmaxf(mx, g[e]); }
        float se = 0.f;
        for (int e = 0; e < 4; ++e) { g[e] = __expf(g[e] - mx); se += g[e]; }
        for (int e = 0; e < 4; ++e) g[e] /= se;
        int i0 = 0;
        for (int e = 1; e < 4; ++e) if (g[e] > g[i0]) i0 = e;
        int i1 = -1;
        for (int e = 0; e < 4; ++e) {
            if (e == i0) continue;
            if (i1 < 0 || g[e] > g[i1]) i1 = e;
        }
        eidx[tid * 2 + 0] = i0;
        eidx[tid * 2 + 1] = i1;
        gval[tid * 2 + 0] = g[i0];
        gval[tid * 2 + 1] = g[i1];
        gsum[tid] = g[i0] + g[i1];
    }
}

// ---------------------------------------------------------------------------
// Kernel 3: fused MoE main kernel.
// Block = (b, 64-pixel tile), 256 threads = 8 wave32.
// LDS tiles are pixel-major so WMMA B-fragments are 2x contiguous b128 loads.
//   y_e = g_e * (P0_e x) * silu(P1_e shared)   (only top-2 experts)
//   z   = P2_e0 y0 + P2_e1 y1 + gsum * x
//   out = proj_out * z
// ---------------------------------------------------------------------------
#define NT  64   // pixels per tile
#define LDX 264  // padded row (256 channels) in bf16 elements, 16B-aligned rows
#define LDY 136  // padded row (128 ranks)

__global__ __launch_bounds__(256) void moe_kernel(
    const float* __restrict__ x, const float* __restrict__ shr,
    const bh* __restrict__ wp0, const bh* __restrict__ wp1,
    const bh* __restrict__ wp2, const bh* __restrict__ wpr,
    const int* __restrict__ eidx, const float* __restrict__ gval,
    const float* __restrict__ gsumA, float* __restrict__ out) {
    __shared__ __align__(16) bh smem[NT * LDX * 2 + NT * LDY * 2];  // 100 KB
    bh* xs  = smem;                    // [NT][LDX] x tile (bf16)
    bh* ssb = smem + NT * LDX;         // [NT][LDX] shared tile
    bh* ys0 = smem + 2 * NT * LDX;     // [NT][LDY] expert-0 y
    bh* ys1 = ys0 + NT * LDY;          // [NT][LDY] expert-1 y
    bh* zs  = ssb;                     // alias: ss dead after Y stage

    int b = blockIdx.y;
    int p0pix = blockIdx.x * NT;
    int tid = threadIdx.x;
    int lane = tid & 31, wave = tid >> 5;
    int half = lane >> 4, lm = lane & 15;

    int e0 = eidx[b * 2], e1 = eidx[b * 2 + 1];
    float g0 = gval[b * 2], g1 = gval[b * 2 + 1];
    float gs = gsumA[b];

    // ---- stage activations to LDS as bf16, pixel-major ----
    const float* xb = x   + (size_t)b * 256 * 16384 + p0pix;
    const float* sb = shr + (size_t)b * 256 * 16384 + p0pix;
    for (int idx = tid; idx < 256 * NT; idx += 256) {
        int c = idx >> 6, n = idx & (NT - 1);
        xs[n * LDX + c]  = (bh)xb[(size_t)c * 16384 + n];
        ssb[n * LDX + c] = (bh)sb[(size_t)c * 16384 + n];
    }
    __syncthreads();

    // ---- Y stage: per expert, y = g * (P0 x) * silu(P1 shared) ----
    for (int ei = 0; ei < 2; ++ei) {
        int e   = ei ? e1 : e0;
        float g = ei ? g1 : g0;
        bh* ysp = ei ? ys1 : ys0;
        const bh* A0 = wp0 + (size_t)e * 128 * 256;
        const bh* A1 = wp1 + (size_t)e * 128 * 256;
        int mt = wave;  // rank tile 0..7
        const bh* a0row = A0 + (mt * 16 + lm) * 256 + half * 8;
        const bh* a1row = A1 + (mt * 16 + lm) * 256 + half * 8;
        for (int nt = 0; nt < 4; ++nt) {
            v8f acc0 = {};
            v8f acc1 = {};
            const bh* brx = xs  + (nt * 16 + lm) * LDX + half * 8;
            const bh* brs = ssb + (nt * 16 + lm) * LDX + half * 8;
#pragma unroll
            for (int ks = 0; ks < 8; ++ks) {
                acc0 = wmma_bf16(frag2(a0row + ks * 32, a0row + ks * 32 + 16),
                                 frag2(brx + ks * 32, brx + ks * 32 + 16), acc0);
                acc1 = wmma_bf16(frag2(a1row + ks * 32, a1row + ks * 32 + 16),
                                 frag2(brs + ks * 32, brs + ks * 32 + 16), acc1);
            }
            int n = nt * 16 + lm;
#pragma unroll
            for (int r = 0; r < 8; ++r) {
                int m = mt * 16 + half * 8 + r;
                float t1 = acc1[r];
                float yv = g * acc0[r] * (t1 / (1.f + __expf(-t1)));
                ysp[n * LDY + m] = (bh)yv;
            }
        }
    }
    __syncthreads();

    // ---- Z stage: z = P2_e0 y0 + P2_e1 y1 + gsum * x ----
    {
        const bh* P2a = wp2 + (size_t)e0 * 256 * 128;
        const bh* P2b = wp2 + (size_t)e1 * 256 * 128;
        for (int i = 0; i < 8; ++i) {
            int tt = wave * 8 + i;
            int mt = tt >> 2, nt = tt & 3;
            v8f acc = {};
            const bh* a0row = P2a + (mt * 16 + lm) * 128 + half * 8;
            const bh* a1row = P2b + (mt * 16 + lm) * 128 + half * 8;
            const bh* b0row = ys0 + (nt * 16 + lm) * LDY + half * 8;
            const bh* b1row = ys1 + (nt * 16 + lm) * LDY + half * 8;
#pragma unroll
            for (int ks = 0; ks < 4; ++ks)
                acc = wmma_bf16(frag2(a0row + ks * 32, a0row + ks * 32 + 16),
                                frag2(b0row + ks * 32, b0row + ks * 32 + 16), acc);
#pragma unroll
            for (int ks = 0; ks < 4; ++ks)
                acc = wmma_bf16(frag2(a1row + ks * 32, a1row + ks * 32 + 16),
                                frag2(b1row + ks * 32, b1row + ks * 32 + 16), acc);
            int n = nt * 16 + lm;
#pragma unroll
            for (int r = 0; r < 8; ++r) {
                int m = mt * 16 + half * 8 + r;
                float v = acc[r] + gs * (float)xs[n * LDX + m];
                zs[n * LDX + m] = (bh)v;
            }
        }
    }
    __syncthreads();

    // ---- OUT stage: out = proj_out * z ----
    {
        float* ob = out + (size_t)b * 256 * 16384 + p0pix;
        for (int i = 0; i < 8; ++i) {
            int tt = wave * 8 + i;
            int mt = tt >> 2, nt = tt & 3;
            v8f acc = {};
            const bh* arow = wpr + (mt * 16 + lm) * 256 + half * 8;
            const bh* brow = zs + (nt * 16 + lm) * LDX + half * 8;
#pragma unroll
            for (int ks = 0; ks < 8; ++ks)
                acc = wmma_bf16(frag2(arow + ks * 32, arow + ks * 32 + 16),
                                frag2(brow + ks * 32, brow + ks * 32 + 16), acc);
            int n = nt * 16 + lm;
#pragma unroll
            for (int r = 0; r < 8; ++r) {
                int m = mt * 16 + half * 8 + r;
                ob[(size_t)m * 16384 + n] = acc[r];
            }
        }
    }
}

// ---------------------------------------------------------------------------
// Host-side launch (graph-capture safe: stream-only, workspace scratch)
// ---------------------------------------------------------------------------
extern "C" void kernel_launch(void* const* d_in, const int* in_sizes, int n_in,
                              void* d_out, int out_size, void* d_ws, size_t ws_size,
                              hipStream_t stream) {
    const float* x      = (const float*)d_in[0];
    const float* shared = (const float*)d_in[1];
    const float* mlp_w1 = (const float*)d_in[2];
    const float* mlp_b1 = (const float*)d_in[3];
    const float* mlp_w2 = (const float*)d_in[4];
    const float* mlp_b2 = (const float*)d_in[5];
    const float* gate_w = (const float*)d_in[6];
    const float* fgate  = (const float*)d_in[7];
    const float* p0     = (const float*)d_in[8];
    const float* p1     = (const float*)d_in[9];
    const float* p2     = (const float*)d_in[10];
    const float* projw  = (const float*)d_in[11];

    char* ws = (char*)d_ws;
    bh*    wp0b = (bh*)(ws + 0);        // 4*128*256 bf16 = 256 KB
    bh*    wp1b = (bh*)(ws + 262144);
    bh*    wp2b = (bh*)(ws + 524288);
    bh*    wprb = (bh*)(ws + 786432);   // 256*256 bf16 = 128 KB
    float* hbuf = (float*)(ws + 917504);
    float* pbuf = (float*)(ws + 925696);
    int*   eidx = (int*)(ws + 933888);
    float* gval = (float*)(ws + 933952);
    float* gsum = (float*)(ws + 934016);

    f2bf_kernel<<<512, 256, 0, stream>>>(p0, wp0b, 131072);
    f2bf_kernel<<<512, 256, 0, stream>>>(p1, wp1b, 131072);
    f2bf_kernel<<<512, 256, 0, stream>>>(p2, wp2b, 131072);
    f2bf_kernel<<<256, 256, 0, stream>>>(projw, wprb, 65536);

    stats_kernel<<<2048, 256, 0, stream>>>(x, hbuf, pbuf);
    route_kernel<<<1, 256, 0, stream>>>(hbuf, pbuf, mlp_w1, mlp_b1, mlp_w2, mlp_b2,
                                        gate_w, fgate, eidx, gval, gsum);

    moe_kernel<<<dim3(256, 8), 256, 0, stream>>>(x, shared, wp0b, wp1b, wp2b, wprb,
                                                 eidx, gval, gsum, (float*)d_out);
}
